// MPNN_21028159881459
// MI455X (gfx1250) — compile-verified, matrix operand order checked by reading, not run on hardware
//
#include <hip/hip_runtime.h>

typedef __attribute__((ext_vector_type(16))) _Float16 v16h;
typedef __attribute__((ext_vector_type(8)))  _Float16 v8h;
typedef __attribute__((ext_vector_type(8)))  float    v8f;
typedef __attribute__((ext_vector_type(4)))  float    f4;

#define DD 32   // embed dim

// ---------------- fragment loaders (ISA 16-bit A layout, wave32) -------------
// Lane l: rows m = l&15 ; K groups {8*(l>>4) .. +7} and {16+8*(l>>4) .. +7}
__device__ __forceinline__ v16h load_a_frag(const float* __restrict__ rowp,
                                            int half, float sc) {
    const float* p = rowp + half * 8;
    f4 x0 = *(const f4*)(p);
    f4 x1 = *(const f4*)(p + 4);
    f4 y0 = *(const f4*)(p + 16);
    f4 y1 = *(const f4*)(p + 20);
    v16h a;
#pragma unroll
    for (int j = 0; j < 4; ++j) {
        a[j]      = (_Float16)(x0[j] * sc);
        a[4 + j]  = (_Float16)(x1[j] * sc);
        a[8 + j]  = (_Float16)(y0[j] * sc);
        a[12 + j] = (_Float16)(y1[j] * sc);
    }
    return a;
}

// B stored transposed as WT[n][k] (n-th output column owns contiguous K),
// so the per-lane K pattern is identical to A: two 16-byte loads.
__device__ __forceinline__ v16h load_b_frag(const _Float16* __restrict__ wrow,
                                            int half) {
    v8h lo = *(const v8h*)(wrow + half * 8);
    v8h hi = *(const v8h*)(wrow + 16 + half * 8);
    v16h b;
#pragma unroll
    for (int j = 0; j < 8; ++j) { b[j] = lo[j]; b[8 + j] = hi[j]; }
    return b;
}

// ---------------- setup kernels ----------------------------------------------
__global__ void mpnn_zero_f32(float* __restrict__ p, int n) {
    int i = blockIdx.x * blockDim.x + threadIdx.x;
    if (i < n) p[i] = 0.0f;
}

__global__ void mpnn_degree(const int* __restrict__ dst, float* __restrict__ cnt, int E) {
    int e = blockIdx.x * blockDim.x + threadIdx.x;
    if (e < E) atomicAdd(&cnt[dst[e]], 1.0f);
}

__global__ void mpnn_invcnt(float* __restrict__ cnt, int N) {  // in-place 1/max(cnt,1)
    int i = blockIdx.x * blockDim.x + threadIdx.x;
    if (i < N) cnt[i] = 1.0f / fmaxf(cnt[i], 1.0f);
}

__global__ void mpnn_init_h(const float* __restrict__ x, const float* __restrict__ nW,
                            const float* __restrict__ nb, float* __restrict__ h, int NF) {
    int i = blockIdx.x * blockDim.x + threadIdx.x;
    if (i < NF) {
        int node = i >> 5, d = i & 31;
        h[i] = fmaf(x[node], nW[d], nb[d]);
    }
}

// WT[n*96 + k] (f16): k<32 -> W1[k][n]=l1_W[k*32+n]; k<64 -> l1_b[(k-32)*32+n];
// else root[(k-64)*32+n]
__global__ void mpnn_prep_w(const float* __restrict__ l1W, const float* __restrict__ l1b,
                            const float* __restrict__ root, _Float16* __restrict__ WT) {
    int i = blockIdx.x * blockDim.x + threadIdx.x;   // 32*96
    if (i < DD * 96) {
        int n = i / 96, k = i % 96;
        float v = (k < 32) ? l1W[k * 32 + n]
                : (k < 64) ? l1b[(k - 32) * 32 + n]
                           : root[(k - 64) * 32 + n];
        WT[i] = (_Float16)v;
    }
}

// ---------------- per-layer kernels ------------------------------------------
// g = relu(h); zero interleaved accumulator SS[node][64] (S0 at +0, S1 at +32)
__global__ void mpnn_relu_zero(const float* __restrict__ h, float* __restrict__ g,
                               float* __restrict__ SS, int NF) {
    int i = blockIdx.x * blockDim.x + threadIdx.x;
    if (i < NF) g[i] = fmaxf(h[i], 0.0f);
    if (i < 2 * NF) SS[i] = 0.0f;
}

// one wave per edge; lane = feature channel. Coalesced 128B row gather, two
// f32 atomics per lane landing in the same 256B region (L2 atomic locality).
__global__ void mpnn_scatter(const int* __restrict__ src, const int* __restrict__ dst,
                             const float* __restrict__ ea, const float* __restrict__ g,
                             float* __restrict__ SS, int E) {
    int idx = blockIdx.x * blockDim.x + threadIdx.x;
    int e = idx >> 5;
    if (e >= E) return;
    int d = idx & 31;
    int s = src[e], t = dst[e];
    float a = ea[e];
    float v = g[s * DD + d];
    float* base = SS + (size_t)t * (2 * DD);
    atomicAdd(base + d, v);            // S0
    atomicAdd(base + DD + d, a * v);   // S1
}

// one wave per 16-node tile: out[16x32] = [ic*S1 | ic*S0 | g] @ WT^T + conv_b
// 2 column tiles x 3 chained v_wmma_f32_16x16x32_f16, f32 accumulate.
// Tail handled by overlapping the last tile (duplicate stores write identical
// values) so EXEC stays all-1s and stores need no per-element guards.
__global__ void __launch_bounds__(256)
mpnn_update_wmma(const float* __restrict__ SS, const float* __restrict__ g,
                 const float* __restrict__ inv_cnt, const _Float16* __restrict__ WT,
                 const float* __restrict__ cb, float* __restrict__ out,
                 int nTiles, int N) {
    int wave = (blockIdx.x * blockDim.x + threadIdx.x) >> 5;  // wave-uniform
    if (wave >= nTiles) return;                               // uniform exit
    int lane = threadIdx.x & 31;
    int half = lane >> 4;
    int r    = lane & 15;
    int base = wave * 16;
    if (base > N - 16) base = N - 16;                         // overlap tail tile
    int row  = base + r;

    float ic = inv_cnt[row];
    const float* ss = SS + (size_t)row * (2 * DD);
    v16h a0 = load_a_frag(ss + DD, half, ic);                 // ic * S1
    v16h a1 = load_a_frag(ss,      half, ic);                 // ic * S0
    v16h a2 = load_a_frag(g + (size_t)row * DD, half, 1.0f);  // g

#pragma unroll
    for (int tc = 0; tc < 2; ++tc) {
        int n = r + tc * 16;
        const _Float16* w = WT + n * 96;
        v16h b0 = load_b_frag(w,      half);   // W1 chunk
        v16h b1 = load_b_frag(w + 32, half);   // B1 chunk
        v16h b2 = load_b_frag(w + 64, half);   // root chunk

        float bias = cb[n];
        v8f c = {bias, bias, bias, bias, bias, bias, bias, bias};
        c = __builtin_amdgcn_wmma_f32_16x16x32_f16(false, a0, false, b0, (short)0, c, false, false);
        c = __builtin_amdgcn_wmma_f32_16x16x32_f16(false, a1, false, b1, (short)0, c, false, false);
        c = __builtin_amdgcn_wmma_f32_16x16x32_f16(false, a2, false, b2, (short)0, c, false, false);

        // C/D layout: VGPR rr -> M = rr + 8*half ; N = (lane&15) + 16*tc
        float* op = out + (size_t)(base + half * 8) * DD + n;
#pragma unroll
        for (int rr = 0; rr < 8; ++rr) op[rr * DD] = c[rr];
    }
}

// ---------------- host launcher ----------------------------------------------
extern "C" void kernel_launch(void* const* d_in, const int* in_sizes, int n_in,
                              void* d_out, int out_size, void* d_ws, size_t ws_size,
                              hipStream_t stream) {
    const float* x    = (const float*)d_in[0];
    const int*   ei   = (const int*)  d_in[1];   // [2,E]: src = ei, dst = ei+E
    const float* ea   = (const float*)d_in[2];
    // d_in[3] = i (==2 -> 3 layers), d_in[4] = dummy : fixed for this problem
    const float* nW   = (const float*)d_in[5];
    const float* nb   = (const float*)d_in[6];
    const float* l1W  = (const float*)d_in[7];
    const float* l1b  = (const float*)d_in[8];
    const float* root = (const float*)d_in[9];
    const float* cb   = (const float*)d_in[10];
    float* out = (float*)d_out;

    const int N  = in_sizes[0];       // 50000
    const int E  = in_sizes[2];       // 250000
    const int NF = N * DD;            // 1.6M floats
    const int nTiles = (N + 15) / 16;

    // workspace carve-up (16B aligned): h + g + SS(2*NF) + cnt + WT ~= 25.8 MB
    float* h   = (float*)d_ws;
    float* g   = h  + NF;
    float* SS  = g  + NF;                       // interleaved [node][S0(32)|S1(32)]
    float* cnt = SS + 2 * (size_t)NF;           // N floats; becomes inv_cnt in place
    _Float16* WT = (_Float16*)(cnt + N);        // 32*96 f16

    const int B = 256;

    // setup
    mpnn_zero_f32<<<(N + B - 1) / B, B, 0, stream>>>(cnt, N);
    mpnn_degree  <<<(E + B - 1) / B, B, 0, stream>>>(ei + E, cnt, E);
    mpnn_invcnt  <<<(N + B - 1) / B, B, 0, stream>>>(cnt, N);
    mpnn_init_h  <<<(NF + B - 1) / B, B, 0, stream>>>(x, nW, nb, h, NF);
    mpnn_prep_w  <<<(DD * 96 + B - 1) / B, B, 0, stream>>>(l1W, l1b, root, WT);

    // 3 encode layers (i=2 -> min(i+1,3)=3); last layer writes d_out directly
    for (int layer = 0; layer < 3; ++layer) {
        mpnn_relu_zero<<<(2 * NF + B - 1) / B, B, 0, stream>>>(h, g, SS, NF);
        mpnn_scatter  <<<((size_t)E * 32 + B - 1) / B, B, 0, stream>>>(
            ei, ei + E, ea, g, SS, E);
        float* dstbuf = (layer == 2) ? out : h;
        mpnn_update_wmma<<<(nTiles * 32 + B - 1) / B, B, 0, stream>>>(
            SS, g, cnt, WT, cb, dstbuf, nTiles, N);
    }
}
